// MultiHeadAttention_9019431321829
// MI455X (gfx1250) — matrix operand order repaired and checked by reading, not compiled, not run on hardware
//
#include <hip/hip_runtime.h>

#define BB 2
#define NN 4096
#define EE 768
#define HH 12
#define HD 64
#define TE 2304            // 3*E
#define MTOT (BB * NN)     // 8192

typedef __bf16 bf16;
typedef __attribute__((ext_vector_type(16))) __bf16 v16bf;
typedef __attribute__((ext_vector_type(8)))  __bf16 v8bf;
typedef __attribute__((ext_vector_type(8)))  float  v8f;

__device__ __forceinline__ bf16 f2bf(float f) {
  unsigned u = __builtin_bit_cast(unsigned, f);
  u += 0x7FFFu + ((u >> 16) & 1u);            // round-to-nearest-even
  unsigned short s = (unsigned short)(u >> 16);
  return __builtin_bit_cast(bf16, s);
}

__device__ __forceinline__ v16bf cat8(v8bf lo, v8bf hi) {
  return __builtin_shufflevector(lo, hi, 0,1,2,3,4,5,6,7,8,9,10,11,12,13,14,15);
}

__device__ __forceinline__ v8f wmma_bf16(v16bf a, v16bf b, v8f c) {
  return __builtin_amdgcn_wmma_f32_16x16x32_bf16(false, a, false, b, (short)0, c,
                                                 false, false);
}

// --- CDNA5 async global->LDS DMA (ASYNCcnt-tracked, no VGPR round trip) ------
__device__ __forceinline__ unsigned lds_off32(const void* p) {
  // generic LDS address: addr[31:0] is the LDS byte offset (ISA aperture rules)
  return (unsigned)(unsigned long long)p;
}
__device__ __forceinline__ void async_load_b128(unsigned lds_off,
                                                const void* gaddr) {
  asm volatile("global_load_async_to_lds_b128 %0, %1, off"
               :: "v"(lds_off), "v"(gaddr) : "memory");
}
__device__ __forceinline__ void wait_async0() {
  asm volatile("s_wait_asynccnt 0x0" ::: "memory");
}

// ---------------------------------------------------------------- convert ---
__global__ void cvt_f32_bf16(const float* __restrict__ in, bf16* __restrict__ out,
                             int n8) {
  int i = blockIdx.x * blockDim.x + threadIdx.x;
  if (i < n8) {
    const float4* p = (const float4*)in + (size_t)i * 2;
    float4 a = p[0], b = p[1];
    v8bf o;
    o[0] = f2bf(a.x); o[1] = f2bf(a.y); o[2] = f2bf(a.z); o[3] = f2bf(a.w);
    o[4] = f2bf(b.x); o[5] = f2bf(b.y); o[6] = f2bf(b.z); o[7] = f2bf(b.w);
    *((v8bf*)out + i) = o;
  }
}

// ------------------------------------------------------- QKV GEMM + scatter ---
// Block: 8 waves = 128 rows x 64 cols. Weight tile (64x32) async-DMA'd into
// double-buffered LDS, shared by all 8 waves.
__global__ void qkv_gemm(const bf16* __restrict__ xb, const bf16* __restrict__ wq,
                         const float* __restrict__ bias,
                         bf16* __restrict__ qb, bf16* __restrict__ kb,
                         bf16* __restrict__ vT) {
  const int t = threadIdx.x;
  const int widx = t >> 5;
  const int lane = t & 31;
  const int g = lane >> 4, ln = lane & 15;
  const int FT = TE / 64;                  // 36 col strips
  const int mb = blockIdx.x / FT, ft = blockIdx.x % FT;
  const int m0 = mb * 128 + widx * 16, f0 = ft * 64;

  __shared__ __align__(64) bf16 wtile[2][64 * 32];   // 8 KB double buffer
  const int sf = t >> 2, sseg = t & 3;               // stage row / 16B segment

  v8f acc[4] = {};
  const bf16* arow  = xb + (m0 + ln) * EE;
  const bf16* wbase = wq + (f0 + sf) * EE + sseg * 8;
  unsigned wl[2] = { lds_off32(&wtile[0][sf * 32 + sseg * 8]),
                     lds_off32(&wtile[1][sf * 32 + sseg * 8]) };

  async_load_b128(wl[0], wbase);                     // prologue DMA
  for (int e0 = 0, s = 0; e0 < EE; e0 += 32, ++s) {
    wait_async0();                                   // my DMA for buf s&1 done
    __syncthreads();                                 // everyone's done
    if (e0 + 32 < EE)                                // DMA next tile under math
      async_load_b128(wl[(s + 1) & 1], wbase + e0 + 32);

    const bf16* wt = &wtile[s & 1][0];
    v16bf a = *(const v16bf*)(arow + e0 + 16 * g);
    v16bf bfr[4];
#pragma unroll
    for (int j = 0; j < 4; ++j)
      bfr[j] = *(const v16bf*)(wt + (j * 16 + ln) * 32 + 16 * g);
#pragma unroll
    for (int j = 0; j < 4; ++j)
      acc[j] = wmma_bf16(a, bfr[j], acc[j]);
  }
#pragma unroll
  for (int j = 0; j < 4; ++j) {
    int f = f0 + j * 16 + ln;
    int h = f / 192, rem = f % 192;
    int c = rem / HD, d = rem % HD;        // c: 0=q 1=k 2=v
    float bv = bias[f];
#pragma unroll
    for (int r = 0; r < 8; ++r) {
      int m = m0 + r + 8 * g;
      int b = m >> 12, nn = m & (NN - 1);
      bf16 o = f2bf(acc[j][r] + bv);
      int bh = b * HH + h;
      if (c == 0)       qb[(bh * NN + nn) * HD + d] = o;
      else if (c == 1)  kb[(bh * NN + nn) * HD + d] = o;
      else              vT[(bh * HD + d) * NN + nn] = o;
    }
  }
}

// ----------------------------------------------------------- flash attention ---
// Block: 8 waves = 8 query tiles of the SAME head; K/V tiles (32 keys) async-
// DMA'd into double-buffered LDS, shared by all waves. Online softmax.
__global__ void flash_attn(const bf16* __restrict__ qb, const bf16* __restrict__ kb,
                           const bf16* __restrict__ vT, bf16* __restrict__ ao) {
  const int t = threadIdx.x;
  const int widx = t >> 5;
  const int lane = t & 31;
  const int g = lane >> 4, ln = lane & 15;
  const int bh = blockIdx.x >> 5;          // 32 query groups per head
  const int qg = blockIdx.x & 31;
  const int h = bh % HH, b = bh / HH;
  const int q0 = (qg * 8 + widx) * 16;

  __shared__ __align__(64) bf16 ktile[2][32 * 64];   // 8 KB
  __shared__ __align__(64) bf16 vtile[2][64 * 32];   // 8 KB
  __shared__ __align__(64) bf16 pbuf[8][16 * 32];    // 8 KB per-wave P staging
  bf16* pb = &pbuf[widx][0];

  const bf16* Q = qb + bh * NN * HD;
  const bf16* K = kb + bh * NN * HD;
  const bf16* V = vT + bh * HD * NN;

  // staging maps (16B per thread per tile)
  const int krow = t >> 3, kseg = t & 7;   // 32 rows x 8 segs
  const int vrow = t >> 2, vseg = t & 3;   // 64 rows x 4 segs
  const bf16* Kst = K + krow * HD + kseg * 8;
  const bf16* Vst = V + vrow * NN + vseg * 8;
  unsigned kl[2] = { lds_off32(&ktile[0][krow * 64 + kseg * 8]),
                     lds_off32(&ktile[1][krow * 64 + kseg * 8]) };
  unsigned vl[2] = { lds_off32(&vtile[0][vrow * 32 + vseg * 8]),
                     lds_off32(&vtile[1][vrow * 32 + vseg * 8]) };

  v16bf qa[2];
  {
    const bf16* qrow = Q + (q0 + ln) * HD;
#pragma unroll
    for (int kk = 0; kk < 2; ++kk) {
      qa[kk] = *(const v16bf*)(qrow + kk * 32 + 16 * g);
    }
  }

  float mrow[8], lrow[8], alpha[8];
#pragma unroll
  for (int r = 0; r < 8; ++r) { mrow[r] = -__builtin_inff(); lrow[r] = 0.f; }
  v8f o[4] = {};
  const float scale = 0.125f;              // 1/sqrt(64)

  async_load_b128(kl[0], Kst);             // prologue DMA
  async_load_b128(vl[0], Vst);
  for (int kt = 0, s = 0; kt < NN; kt += 32, ++s) {
    wait_async0();
    __syncthreads();
    if (kt + 32 < NN) {                    // DMA next K/V tiles under math
      async_load_b128(kl[(s + 1) & 1], Kst + (kt + 32) * HD);
      async_load_b128(vl[(s + 1) & 1], Vst + (kt + 32));
    }
    const bf16* kt_lds = &ktile[s & 1][0];
    const bf16* vt_lds = &vtile[s & 1][0];

    v8f s0 = {}, s1 = {};
    {
      v16bf kf[4];
#pragma unroll
      for (int kk = 0; kk < 2; ++kk) {
        kf[kk * 2]     = *(const v16bf*)(kt_lds + ln * 64 + kk * 32 + 16 * g);
        kf[kk * 2 + 1] = *(const v16bf*)(kt_lds + (16 + ln) * 64 + kk * 32 + 16 * g);
      }
#pragma unroll
      for (int kk = 0; kk < 2; ++kk) {
        s0 = wmma_bf16(qa[kk], kf[kk * 2], s0);
        s1 = wmma_bf16(qa[kk], kf[kk * 2 + 1], s1);
      }
    }
    s0 = s0 * scale; s1 = s1 * scale;
#pragma unroll
    for (int r = 0; r < 8; ++r) {
      float tm = fmaxf(s0[r], s1[r]);
      tm = fmaxf(tm, __shfl_xor(tm, 1));
      tm = fmaxf(tm, __shfl_xor(tm, 2));
      tm = fmaxf(tm, __shfl_xor(tm, 4));
      tm = fmaxf(tm, __shfl_xor(tm, 8));
      float mn = fmaxf(mrow[r], tm);
      alpha[r] = __expf(mrow[r] - mn);
      mrow[r] = mn;
      float p0 = __expf(s0[r] - mn);
      float p1 = __expf(s1[r] - mn);
      float ps = p0 + p1;
      ps += __shfl_xor(ps, 1);
      ps += __shfl_xor(ps, 2);
      ps += __shfl_xor(ps, 4);
      ps += __shfl_xor(ps, 8);
      lrow[r] = lrow[r] * alpha[r] + ps;
      pb[(r + 8 * g) * 32 + ln] = f2bf(p0);
      pb[(r + 8 * g) * 32 + 16 + ln] = f2bf(p1);
    }
#pragma unroll
    for (int j = 0; j < 4; ++j)
#pragma unroll
      for (int r = 0; r < 8; ++r) o[j][r] *= alpha[r];

    // Re-layout P (C-layout in LDS) into A-fragment; in-order DS within a wave.
    v16bf pa = *(const v16bf*)(pb + ln * 32 + 16 * g);
    v16bf vf[4];
#pragma unroll
    for (int j = 0; j < 4; ++j)
      vf[j] = *(const v16bf*)(vt_lds + (j * 16 + ln) * 32 + 16 * g);
#pragma unroll
    for (int j = 0; j < 4; ++j)
      o[j] = wmma_bf16(pa, vf[j], o[j]);
  }

#pragma unroll
  for (int j = 0; j < 4; ++j)
#pragma unroll
    for (int r = 0; r < 8; ++r) {
      int row = q0 + r + 8 * g;
      int e = h * HD + j * 16 + ln;
      ao[(b * NN + row) * EE + e] = f2bf(o[j][r] / lrow[r]);
    }
}

// -------------------------------------------------------- output projection ---
__global__ void proj_gemm(const bf16* __restrict__ ab, const bf16* __restrict__ wp,
                          const float* __restrict__ bias, float* __restrict__ out) {
  const int t = threadIdx.x;
  const int widx = t >> 5;
  const int lane = t & 31;
  const int g = lane >> 4, ln = lane & 15;
  const int FT = EE / 64;                  // 12 col strips
  const int mb = blockIdx.x / FT, ft = blockIdx.x % FT;
  const int m0 = mb * 128 + widx * 16, f0 = ft * 64;

  __shared__ __align__(64) bf16 wtile[2][64 * 32];
  const int sf = t >> 2, sseg = t & 3;

  v8f acc[4] = {};
  const bf16* arow  = ab + (m0 + ln) * EE;
  const bf16* wbase = wp + (f0 + sf) * EE + sseg * 8;
  unsigned wl[2] = { lds_off32(&wtile[0][sf * 32 + sseg * 8]),
                     lds_off32(&wtile[1][sf * 32 + sseg * 8]) };

  async_load_b128(wl[0], wbase);
  for (int e0 = 0, s = 0; e0 < EE; e0 += 32, ++s) {
    wait_async0();
    __syncthreads();
    if (e0 + 32 < EE)
      async_load_b128(wl[(s + 1) & 1], wbase + e0 + 32);

    const bf16* wt = &wtile[s & 1][0];
    v16bf a = *(const v16bf*)(arow + e0 + 16 * g);
    v16bf bfr[4];
#pragma unroll
    for (int j = 0; j < 4; ++j)
      bfr[j] = *(const v16bf*)(wt + (j * 16 + ln) * 32 + 16 * g);
#pragma unroll
    for (int j = 0; j < 4; ++j)
      acc[j] = wmma_bf16(a, bfr[j], acc[j]);
  }
#pragma unroll
  for (int j = 0; j < 4; ++j) {
    float bv = bias[f0 + j * 16 + ln];
#pragma unroll
    for (int r = 0; r < 8; ++r) {
      int m = m0 + r + 8 * g;
      out[m * EE + f0 + j * 16 + ln] = acc[j][r] + bv;
    }
  }
}

// ------------------------------------------------------------------ launch ---
extern "C" void kernel_launch(void* const* d_in, const int* in_sizes, int n_in,
                              void* d_out, int out_size, void* d_ws, size_t ws_size,
                              hipStream_t stream) {
  const float* x      = (const float*)d_in[0];
  const float* w_qkv  = (const float*)d_in[1];
  const float* b_qkv  = (const float*)d_in[2];
  const float* w_proj = (const float*)d_in[3];
  const float* b_proj = (const float*)d_in[4];
  float* out = (float*)d_out;

  char* ws = (char*)d_ws;
  size_t off = 0;
  auto alloc = [&](size_t bytes) -> void* {
    void* p = ws + off;
    off += (bytes + 255) & ~(size_t)255;
    return p;
  };
  bf16* xb  = (bf16*)alloc((size_t)MTOT * EE * 2);
  bf16* wqb = (bf16*)alloc((size_t)TE * EE * 2);
  bf16* wpb = (bf16*)alloc((size_t)EE * EE * 2);
  bf16* qb  = (bf16*)alloc((size_t)BB * HH * NN * HD * 2);
  bf16* kb  = (bf16*)alloc((size_t)BB * HH * NN * HD * 2);
  bf16* vT  = (bf16*)alloc((size_t)BB * HH * NN * HD * 2);
  bf16* ao  = (bf16*)alloc((size_t)MTOT * EE * 2);

  int n8 = MTOT * EE / 8;
  cvt_f32_bf16<<<(n8 + 255) / 256, 256, 0, stream>>>(x, xb, n8);
  n8 = TE * EE / 8;
  cvt_f32_bf16<<<(n8 + 255) / 256, 256, 0, stream>>>(w_qkv, wqb, n8);
  n8 = EE * EE / 8;
  cvt_f32_bf16<<<(n8 + 255) / 256, 256, 0, stream>>>(w_proj, wpb, n8);

  // (M/128)*(3E/64) = 64*36 blocks, 8 waves each
  qkv_gemm<<<64 * 36, 256, 0, stream>>>(xb, wqb, b_qkv, qb, kb, vT);
  // B*H*(N/(16*8)) = 24*32 blocks
  flash_attn<<<24 * 32, 256, 0, stream>>>(qb, kb, vT, ao);
  // (M/128)*(E/64) = 64*12 blocks
  proj_gemm<<<64 * 12, 256, 0, stream>>>(ao, wpb, b_proj, out);
}
